// SparseUResNet_65541200937358
// MI455X (gfx1250) — compile-verified, matrix operand order checked by reading, not run on hardware
//
#include <hip/hip_runtime.h>
#include <cstdint>
#include <cstddef>

#define HWDIM 512
#define BATCH 4
#define EPSV 1e-4f

typedef __attribute__((ext_vector_type(16))) _Float16 v16h;
typedef __attribute__((ext_vector_type(8)))  float    v8f;
typedef __attribute__((ext_vector_type(4)))  unsigned v4u;
typedef __attribute__((ext_vector_type(4)))  int      v4i;

#if defined(__has_builtin)
#  if __has_builtin(__builtin_amdgcn_global_load_async_to_lds_b128)
#    define USE_ASYNC_LDS 1
#  endif
#endif
#ifndef USE_ASYNC_LDS
#  define USE_ASYNC_LDS 0
#endif

// ---------------------------------------------------------------------------
// Fragment loader for V_WMMA_F32_16X16X32_F16 (wave32 layout, ISA 7.12.2).
// For one lane the 16 halfs are two contiguous 16B chunks:
//   lanes 0-15 : halfs [0,8)   and [16,24)   (K 0..7, 16..23)
//   lanes 16-31: halfs [8,16)  and [24,32)   (K 8..15, 24..31)
// => two b128 loads (ds_load_b128 from LDS / global_load_b128 for weights).
// B (32x16, N=lane%16) mirrors this since weights are staged [tap][co][ci].
// ---------------------------------------------------------------------------
static __device__ __forceinline__ v16h load_frag16(const _Float16* base) {
  union { v16h v; v4u q[2]; } f;
  const _Float16* p = base + ((threadIdx.x & 16) ? 8 : 0);
  f.q[0] = *(const v4u*)(p);
  f.q[1] = *(const v4u*)(p + 16);
  return f.v;
}

// ---------------------------------------------------------------------------
// Generic dense conv as implicit GEMM.
//   in  : f16 activations [B,Hi,Wi,Cip]   (already bn/relu/masked, zero-padded)
//   wT  : f16 weights     [KH*KW,Cop,Cip] (transposed + zero-padded, L2-resident)
//   out : f32             [B,Ho,Wo,Co]
// Epilogue fuses subconv's mask-multiply and the residual skip-add.
// Block = 256 threads = 8 waves: 4 pixel sub-tiles (16 px) x 2 channel
// sub-tiles (16 ch) => 8x8 pixels x 32 channels per block.
// Input patch staged in LDS via CDNA5 async global->LDS copies when available.
// ---------------------------------------------------------------------------
__global__ __launch_bounds__(256) void conv_wmma_kernel(
    const _Float16* __restrict__ in, const _Float16* __restrict__ wT,
    const float* __restrict__ mask, const float* __restrict__ skip,
    float* __restrict__ out,
    int Hi, int Wi, int Cip, int Ho, int Wo, int Co, int Cop,
    int KH, int KW, int stride, int pad)
{
  extern __shared__ char smem_raw[];
  _Float16* patch = (_Float16*)smem_raw;

  const int PH = 7 * stride + KH, PW = 7 * stride + KW;
  const int tilesX = Wo >> 3, tilesY = Ho >> 3;
  int t = blockIdx.x;
  const int b = t / (tilesX * tilesY); t -= b * tilesX * tilesY;
  const int ty = t / tilesX, tx = t - ty * tilesX;
  const int iy0 = ty * 8 * stride - pad, ix0 = tx * 8 * stride - pad;

  // Cooperative patch fill, 16B granules (Cip is a multiple of 32).
  const int c8 = Cip >> 3;
  const int nvec = PH * PW * c8;
  for (int i = threadIdx.x; i < nvec; i += 256) {
    const int pc = i % c8;
    int pix = i / c8;
    const int py = pix / PW, px = pix - (pix / PW) * PW;
    const int iy = iy0 + py, ix = ix0 + px;
    const bool inb = (iy >= 0 && iy < Hi && ix >= 0 && ix < Wi);
    _Float16* dst = patch + (size_t)i * 8;
    const _Float16* src = in + (((size_t)b * Hi + iy) * Wi + ix) * Cip + pc * 8;
#if USE_ASYNC_LDS
    if (inb) {
      __builtin_amdgcn_global_load_async_to_lds_b128(
          (__attribute__((address_space(1))) v4i*)src,
          (__attribute__((address_space(3))) v4i*)dst, 0, 0);
    } else {
      v4u z = {0u, 0u, 0u, 0u};
      *(v4u*)dst = z;
    }
#else
    v4u val = {0u, 0u, 0u, 0u};
    if (inb) val = *(const v4u*)src;
    *(v4u*)dst = val;
#endif
  }
#if USE_ASYNC_LDS
  asm volatile("s_wait_asynccnt 0" ::: "memory");
#endif
  __syncthreads();

  const int lane = threadIdx.x & 31;
  const int wave = threadIdx.x >> 5;
  const int sm = wave & 3, sn = wave >> 2;
  const int m  = lane & 15;
  const int p  = sm * 16 + m, oy = p >> 3, ox = p & 7;
  const int ncol = (blockIdx.y << 5) + (sn << 4) + m;

  v8f acc = {};
  const int ntap = KH * KW;
  for (int tap = 0; tap < ntap; ++tap) {
    const int dy = tap / KW, dx = tap - (tap / KW) * KW;
    const _Float16* arow = patch + (size_t)((oy * stride + dy) * PW + (ox * stride + dx)) * Cip;
    const _Float16* bcol = wT + ((size_t)tap * Cop + ncol) * Cip;
    for (int kc = 0; kc < Cip; kc += 32) {
      v16h a = load_frag16(arow + kc);
      v16h bf = load_frag16(bcol + kc);
      acc = __builtin_amdgcn_wmma_f32_16x16x32_f16(false, a, false, bf,
                                                   (short)0, acc, false, false);
    }
  }

  if (ncol < Co) {
    const int rb = (threadIdx.x & 16) ? 8 : 0;   // C/D layout: M = vgpr + 8*(lane>=16)
#pragma unroll
    for (int r = 0; r < 8; ++r) {
      const int pp = sm * 16 + r + rb;
      const int gy = ty * 8 + (pp >> 3), gx = tx * 8 + (pp & 7);
      const size_t pix = ((size_t)b * Ho + gy) * Wo + gx;
      float v = acc[r];
      if (mask) v *= mask[pix];
      if (skip) v += skip[pix * Co + ncol];
      out[pix * Co + ncol] = v;
    }
  }
}

// ---------------------------------------------------------------------------
// Transposed 2x2 up-conv (einsum bhwc,ijcd->bhiwjd):
// out[b,2h+i,2w+j,d] = sum_c in[b,h,w,c] * W[i,j,c,d].
// Each wave owns one output parity (i,j) == one tap, so B is tap-uniform.
// 16 pixels per wave = 4x4 input tile; block covers 8x8 output pixels.
// ---------------------------------------------------------------------------
__global__ __launch_bounds__(256) void upconv_wmma_kernel(
    const _Float16* __restrict__ in, const _Float16* __restrict__ wT, // [4,Cop,Cip]
    const float* __restrict__ mask, float* __restrict__ out,
    int Hi, int Wi, int Cip, int Co, int Cop)
{
  const int tilesX = Wi >> 2, tilesY = Hi >> 2;
  int t = blockIdx.x;
  const int b = t / (tilesX * tilesY); t -= b * tilesX * tilesY;
  const int ty = t / tilesX, tx = t - ty * tilesX;

  const int lane = threadIdx.x & 31;
  const int wave = threadIdx.x >> 5;
  const int tap = wave & 3, sn = wave >> 2;
  const int di = tap >> 1, dj = tap & 1;
  const int m = lane & 15;
  const int iy = ty * 4 + (m >> 2), ix = tx * 4 + (m & 3);
  const int ncol = (blockIdx.y << 5) + (sn << 4) + m;

  const _Float16* arow = in + (((size_t)b * Hi + iy) * Wi + ix) * Cip;
  const _Float16* bcol = wT + ((size_t)tap * Cop + ncol) * Cip;

  v8f acc = {};
  for (int kc = 0; kc < Cip; kc += 32) {
    v16h a = load_frag16(arow + kc);
    v16h bf = load_frag16(bcol + kc);
    acc = __builtin_amdgcn_wmma_f32_16x16x32_f16(false, a, false, bf,
                                                 (short)0, acc, false, false);
  }

  if (ncol < Co) {
    const int rb = (threadIdx.x & 16) ? 8 : 0;
    const int Ho = Hi << 1, Wo = Wi << 1;
#pragma unroll
    for (int r = 0; r < 8; ++r) {
      const int mm = r + rb;
      const int oy = (ty * 4 + (mm >> 2)) * 2 + di;
      const int ox = (tx * 4 + (mm & 3)) * 2 + dj;
      const size_t pix = ((size_t)b * Ho + oy) * Wo + ox;
      out[pix * Co + ncol] = acc[r] * (mask ? mask[pix] : 1.0f);
    }
  }
}

// ----------------------------- elementwise ---------------------------------
__global__ void fill_u32_kernel(unsigned* p, unsigned v, size_t n) {
  size_t i = (size_t)blockIdx.x * 256 + threadIdx.x;
  if (i < n) p[i] = v;
}

__global__ void scatter_kernel(const int* __restrict__ coords,
                               const float* __restrict__ feat,
                               _Float16* __restrict__ dense,
                               float* __restrict__ mask, int npts) {
  int i = blockIdx.x * 256 + threadIdx.x;
  if (i >= npts) return;
  const int r = coords[3 * i], c = coords[3 * i + 1], b = coords[3 * i + 2];
  const size_t pix = ((size_t)b * HWDIM + r) * HWDIM + c;
  dense[pix * 32] = (_Float16)feat[i];   // Cin=1 padded to 32
  mask[pix] = 1.0f;
}

// fused (optional BN+ReLU) + mask + f32->f16 cast with channel zero-padding
__global__ void bncast_f16_kernel(const float* __restrict__ x,
                                  const float* __restrict__ bias,
                                  const float* __restrict__ mean,
                                  const float* __restrict__ scale,
                                  const float* __restrict__ var,
                                  const float* __restrict__ mask,
                                  _Float16* __restrict__ y,
                                  int C, int Cip, size_t npix) {
  size_t i = (size_t)blockIdx.x * 256 + threadIdx.x;
  const size_t total = npix * (size_t)Cip;
  if (i >= total) return;
  const int c = (int)(i % Cip);
  const size_t pix = i / Cip;
  float v = 0.0f;
  if (c < C) {
    v = x[pix * C + c];
    if (scale) {
      const float s = scale[c] * rsqrtf(var[c] + EPSV);
      v = (v - mean[c]) * s + bias[c];
      v = v > 0.0f ? v : 0.0f;
    }
    if (mask) v *= mask[pix];
  }
  y[i] = (_Float16)v;
}

// [kh,kw,ci,co] f32  ->  [tap,co,ci] f16, zero-padded to Cop/Cip
__global__ void wconv_kernel(const float* __restrict__ w, _Float16* __restrict__ wT,
                             int KHKW, int Ci, int Co, int Cip, int Cop) {
  size_t i = (size_t)blockIdx.x * 256 + threadIdx.x;
  const size_t total = (size_t)KHKW * Cop * Cip;
  if (i >= total) return;
  const int tap = (int)(i / ((size_t)Cop * Cip));
  const int r = (int)(i - (size_t)tap * Cop * Cip);
  const int co = r / Cip, ci = r - (r / Cip) * Cip;
  float v = 0.0f;
  if (co < Co && ci < Ci) v = w[((size_t)tap * Ci + ci) * Co + co];
  wT[i] = (_Float16)v;
}

__global__ void maskdown_kernel(const float* __restrict__ m, float* __restrict__ m2,
                                int H, int W) {
  const int H2 = H >> 1, W2 = W >> 1;
  const size_t tot = (size_t)BATCH * H2 * W2;
  size_t i = (size_t)blockIdx.x * 256 + threadIdx.x;
  if (i >= tot) return;
  const int x = (int)(i % W2);
  size_t r = i / W2;
  const int y = (int)(r % H2);
  const int b = (int)(r / H2);
  const float* base = m + (((size_t)b * H + 2 * y) * W + 2 * x);
  m2[i] = fmaxf(fmaxf(base[0], base[1]), fmaxf(base[W], base[W + 1]));
}

__global__ void concat_kernel(const float* __restrict__ a, const float* __restrict__ bsrc,
                              float* __restrict__ o, int C, size_t npix) {
  const size_t tot = npix * (size_t)(2 * C);
  size_t i = (size_t)blockIdx.x * 256 + threadIdx.x;
  if (i >= tot) return;
  const int c = (int)(i % (2 * C));
  const size_t pix = i / (2 * C);
  o[i] = (c < C) ? a[pix * C + c] : bsrc[pix * C + (c - C)];
}

__global__ void gather_softmax_kernel(const int* __restrict__ coords,
                                      const float* __restrict__ logits,
                                      float* __restrict__ out, int npts) {
  int i = blockIdx.x * 256 + threadIdx.x;
  if (i >= npts) return;
  const int r = coords[3 * i], c = coords[3 * i + 1], b = coords[3 * i + 2];
  const size_t pix = ((size_t)b * HWDIM + r) * HWDIM + c;
  const float l0 = logits[pix * 3 + 0], l1 = logits[pix * 3 + 1], l2 = logits[pix * 3 + 2];
  const float mx = fmaxf(l0, fmaxf(l1, l2));
  const float e0 = expf(l0 - mx), e1 = expf(l1 - mx), e2 = expf(l2 - mx);
  const float inv = 1.0f / (e0 + e1 + e2);
  out[3 * i + 0] = e0 * inv;
  out[3 * i + 1] = e1 * inv;
  out[3 * i + 2] = e2 * inv;
}

// ============================ host-side driver =============================
namespace {

struct Bump {
  char* base; size_t off, cap;
  void* take(size_t bytes) {
    void* r = base + off;
    off = (off + bytes + 255) & ~(size_t)255;
    return r;
  }
};

struct BN  { const float *bias, *mean, *scale, *var; };
struct Blk { BN bn1, bn2; const float *c1, *c2, *nin; };
struct ULevel {
  bool inner;
  BN down_bn; const float* down_w;
  Blk post[2], pre[2];
  BN up_bn; const float* up_w;
};

struct Walker {
  void* const* d_in; int idx;
  const float* next() { return (const float*)d_in[idx++]; }
};

// jax pytree flattening: dict keys sorted, lists in order, depth-first.
BN readBN(Walker& w) {
  BN b;
  b.bias = w.next(); b.mean = w.next(); b.scale = w.next(); b.var = w.next();
  return b;
}
Blk readBlk(Walker& w, bool has_nin) {
  Blk k;
  k.bn1 = readBN(w); k.bn2 = readBN(w);
  k.c1 = w.next(); k.c2 = w.next();
  k.nin = has_nin ? w.next() : nullptr;
  return k;
}
void readULevel(Walker& w, ULevel* lv, int d, int nlev) {
  ULevel& L = lv[d];
  L.inner = (d + 1 < nlev);
  if (L.inner) {  // sorted: down_bn, down_w, inner, post, pre, up_bn, up_w
    L.down_bn = readBN(w); L.down_w = w.next();
    readULevel(w, lv, d + 1, nlev);
    L.post[0] = readBlk(w, true);
    L.post[1] = readBlk(w, false);
    L.pre[0] = readBlk(w, false);
    L.pre[1] = readBlk(w, false);
    L.up_bn = readBN(w); L.up_w = w.next();
  } else {        // sorted: pre
    L.pre[0] = readBlk(w, false);
    L.pre[1] = readBlk(w, false);
  }
}

inline int al32(int c) { return (c + 31) & ~31; }

struct Ctx { Bump ws; hipStream_t s; };

const _Float16* stage_w(Ctx& ct, const float* w, int KH, int KW, int Ci, int Co) {
  const int Cip = al32(Ci), Cop = al32(Co);
  const size_t n = (size_t)KH * KW * Cop * Cip;
  _Float16* wT = (_Float16*)ct.ws.take(n * 2);
  wconv_kernel<<<(unsigned)((n + 255) / 256), 256, 0, ct.s>>>(w, wT, KH * KW, Ci, Co, Cip, Cop);
  return wT;
}

void conv(Ctx& ct, const _Float16* in, const _Float16* wT, const float* mask,
          const float* skip, float* out, int Hi, int Wi, int Ci,
          int Ho, int Wo, int Co, int KH, int KW, int stride, int pad) {
  const int Cip = al32(Ci), Cop = al32(Co);
  dim3 grid((unsigned)((size_t)BATCH * (Ho / 8) * (Wo / 8)), (unsigned)(Cop / 32));
  const int PH = 7 * stride + KH, PW = 7 * stride + KW;
  const size_t lds = (size_t)PH * PW * Cip * 2;
  conv_wmma_kernel<<<grid, 256, lds, ct.s>>>(in, wT, mask, skip, out,
                                             Hi, Wi, Cip, Ho, Wo, Co, Cop,
                                             KH, KW, stride, pad);
}

_Float16* bncast(Ctx& ct, const float* x, const BN* bn, const float* mask,
                 int C, int H, int W) {
  const int Cip = al32(C);
  const size_t npix = (size_t)BATCH * H * W, n = npix * Cip;
  _Float16* y = (_Float16*)ct.ws.take(n * 2);
  bncast_f16_kernel<<<(unsigned)((n + 255) / 256), 256, 0, ct.s>>>(
      x, bn ? bn->bias : nullptr, bn ? bn->mean : nullptr,
      bn ? bn->scale : nullptr, bn ? bn->var : nullptr,
      mask, y, C, Cip, npix);
  return y;
}

float* run_block(Ctx& ct, const Blk& bk, int a, int b, float* x,
                 const float* mask, int H, int W) {
  const size_t npix = (size_t)BATCH * H * W;
  const float* skip = x;
  if (bk.nin) {
    _Float16* xh = bncast(ct, x, nullptr, nullptr, a, H, W);  // plain cast
    const _Float16* ninT = stage_w(ct, bk.nin, 1, 1, a, b);
    float* sk = (float*)ct.ws.take(npix * b * 4);
    conv(ct, xh, ninT, nullptr, nullptr, sk, H, W, a, H, W, b, 1, 1, 1, 0);
    skip = sk;
  }
  _Float16* h1 = bncast(ct, x, &bk.bn1, mask, a, H, W);
  const _Float16* w1T = stage_w(ct, bk.c1, 3, 3, a, b);
  float* y1 = (float*)ct.ws.take(npix * b * 4);
  conv(ct, h1, w1T, mask, nullptr, y1, H, W, a, H, W, b, 3, 3, 1, 1);
  _Float16* h2 = bncast(ct, y1, &bk.bn2, mask, b, H, W);
  const _Float16* w2T = stage_w(ct, bk.c2, 3, 3, b, b);
  float* y2 = (float*)ct.ws.take(npix * b * 4);
  conv(ct, h2, w2T, mask, skip, y2, H, W, b, H, W, b, 3, 3, 1, 1);
  return y2;
}

const int PLANES[5] = {16, 32, 48, 64, 80};

float* unet_run(Ctx& ct, ULevel* lv, int d, float* x, float* mask, int H, int W) {
  ULevel& L = lv[d];
  const int P = PLANES[d];
  for (int i = 0; i < 2; ++i) x = run_block(ct, L.pre[i], P, P, x, mask, H, W);
  if (L.inner) {
    const int P2 = PLANES[d + 1];
    float* skip = x;
    _Float16* xh = bncast(ct, x, &L.down_bn, mask, P, H, W);
    const _Float16* dwT = stage_w(ct, L.down_w, 2, 2, P, P2);
    const int H2 = H / 2, W2 = W / 2;
    const size_t npix2 = (size_t)BATCH * H2 * W2;
    float* y = (float*)ct.ws.take(npix2 * P2 * 4);
    conv(ct, xh, dwT, nullptr, nullptr, y, H, W, P, H2, W2, P2, 2, 2, 2, 0);
    float* mask2 = (float*)ct.ws.take(npix2 * 4);
    maskdown_kernel<<<(unsigned)((npix2 + 255) / 256), 256, 0, ct.s>>>(mask, mask2, H, W);
    y = unet_run(ct, lv, d + 1, y, mask2, H2, W2);
    _Float16* yh = bncast(ct, y, &L.up_bn, mask2, P2, H2, W2);
    const _Float16* uwT = stage_w(ct, L.up_w, 2, 2, P2, P);
    const size_t npix = (size_t)BATCH * H * W;
    float* yup = (float*)ct.ws.take(npix * P * 4);
    dim3 ug((unsigned)((size_t)BATCH * (H2 / 4) * (W2 / 4)), (unsigned)(al32(P) / 32));
    upconv_wmma_kernel<<<ug, 256, 0, ct.s>>>(yh, uwT, mask, yup, H2, W2,
                                             al32(P2), P, al32(P));
    float* xc = (float*)ct.ws.take(npix * 2 * P * 4);
    const size_t tot = npix * 2 * P;
    concat_kernel<<<(unsigned)((tot + 255) / 256), 256, 0, ct.s>>>(skip, yup, xc, P, npix);
    x = xc;
    x = run_block(ct, L.post[0], 2 * P, P, x, mask, H, W);
    x = run_block(ct, L.post[1], P, P, x, mask, H, W);
  }
  return x;
}

} // namespace

extern "C" void kernel_launch(void* const* d_in, const int* in_sizes, int n_in,
                              void* d_out, int out_size, void* d_ws, size_t ws_size,
                              hipStream_t stream) {
  const int* coords = (const int*)d_in[0];
  const float* feat = (const float*)d_in[1];
  const int npts = in_sizes[0] / 3;

  Walker w{d_in, 3};                 // params pytree starts after coord/feat/batchsize
  BN bn_out = readBN(w);             // 'bn_out'
  ULevel levels[5];
  readULevel(w, levels, 0, 5);       // 'unet'
  const float* w_in = w.next();      // 'w_in'
  const float* w_out = w.next();     // 'w_out'

  Ctx ct{Bump{(char*)d_ws, 0, ws_size}, stream};

  const size_t npix = (size_t)BATCH * HWDIM * HWDIM;
  _Float16* dense = (_Float16*)ct.ws.take(npix * 32 * 2);
  float* mask = (float*)ct.ws.take(npix * 4);
  const size_t zn = npix * 32 / 2;   // halfs -> u32 words
  fill_u32_kernel<<<(unsigned)((zn + 255) / 256), 256, 0, stream>>>((unsigned*)dense, 0u, zn);
  fill_u32_kernel<<<(unsigned)((npix + 255) / 256), 256, 0, stream>>>((unsigned*)mask, 0u, npix);
  scatter_kernel<<<(unsigned)((npts + 255) / 256), 256, 0, stream>>>(coords, feat, dense, mask, npts);

  const _Float16* winT = stage_w(ct, w_in, 3, 3, 1, 16);
  float* x = (float*)ct.ws.take(npix * 16 * 4);
  conv(ct, dense, winT, mask, nullptr, x, HWDIM, HWDIM, 1, HWDIM, HWDIM, 16, 3, 3, 1, 1);

  x = unet_run(ct, levels, 0, x, mask, HWDIM, HWDIM);

  _Float16* xh = bncast(ct, x, &bn_out, mask, 16, HWDIM, HWDIM);
  const _Float16* woutT = stage_w(ct, w_out, 1, 1, 16, 3);
  float* logits = (float*)ct.ws.take(npix * 3 * 4);
  conv(ct, xh, woutT, nullptr, nullptr, logits, HWDIM, HWDIM, 16, HWDIM, HWDIM, 3, 1, 1, 1, 0);

  gather_softmax_kernel<<<(unsigned)((npts + 255) / 256), 256, 0, stream>>>(
      coords, logits, (float*)d_out, npts);
}
